// GraphAttention_29240137351217
// MI455X (gfx1250) — matrix operand order repaired and checked by reading, not compiled, not run on hardware
//
#include <hip/hip_runtime.h>
#include <hip/hip_bf16.h>

typedef __attribute__((ext_vector_type(16))) _Float16 v16h;
typedef __attribute__((ext_vector_type(8)))  _Float16 v8h;
typedef __attribute__((ext_vector_type(8)))  float    v8f;
typedef __attribute__((ext_vector_type(4)))  float    v4f;
typedef __attribute__((ext_vector_type(2)))  float    v2f;

#define BS   8
#define NN   4096
#define FIN  128
#define FOUT 128
#define NEG_SLOPE 0.2f

__device__ __forceinline__ float lrelu(float x) {
    // leaky_relu(x) == max(x, 0.2*x) for slope in (0,1)
    return fmaxf(x, NEG_SLOPE * x);
}

// ---------------------------------------------------------------------------
// Kernel 1: h_prime = h @ w  (fp32 WMMA 16x16x4 for full precision)
// Writes h_prime as f16 TRANSPOSED: hpT[b][f][j]  (j contiguous)  -> this is
// exactly the B-fragment-friendly layout for the 16x16x32 f16 WMMA later.
// Also computes s_i = h_prime[i] . a_src and d_i = h_prime[i] . a_dst.
// One wave = 16 rows x 128 cols. Block = 4 waves = 64 rows.
// ---------------------------------------------------------------------------
__global__ void __launch_bounds__(128)
gat_k1_hprime(const float* __restrict__ h, const float* __restrict__ w,
              const float* __restrict__ a_src, const float* __restrict__ a_dst,
              _Float16* __restrict__ hpT, float* __restrict__ s_out,
              float* __restrict__ d_out_)
{
    const int lane  = threadIdx.x & 31;
    const int wave  = threadIdx.x >> 5;
    const int l15   = lane & 15;
    const int halfL = lane >> 4;

    const int tile  = blockIdx.x * 4 + wave;  // 16-row tile index, global
    const int row0  = tile * 16;              // global row in [0, BS*NN)
    const int b     = row0 / NN;
    const int i0    = row0 % NN;

    v8f C[8] = {};  // 16x128 fp32 accumulators (8 N-tiles of 16)

    // K loop, 4 per WMMA. A fragment (f32 16x4): lane holds M=l15, K=halfL*2+{0,1}
    const float* arow = h + (size_t)(row0 + l15) * FIN;
    for (int k = 0; k < FIN; k += 4) {
        v2f A = *(const v2f*)(arow + k + halfL * 2);
        #pragma unroll
        for (int t = 0; t < 8; ++t) {
            // B fragment (f32 4x16): lane holds N=l15, K=halfL*2+{0,1}
            v2f B;
            B.x = w[(k + halfL * 2 + 0) * FOUT + 16 * t + l15];
            B.y = w[(k + halfL * 2 + 1) * FOUT + 16 * t + l15];
            C[t] = __builtin_amdgcn_wmma_f32_16x16x4_f32(
                false, A, false, B, (short)0, C[t], false, false);
        }
    }

    // C fragment layout: element r of C[t] = h_prime[i0 + r + halfL*8][16t + l15]
    // Store transposed f16: hpT[(b*FOUT + f)*NN + j], j contiguous over r.
    #pragma unroll
    for (int t = 0; t < 8; ++t) {
        v8h pk;
        #pragma unroll
        for (int r = 0; r < 8; ++r) pk[r] = (_Float16)C[t][r];
        _Float16* dst = hpT + (size_t)(b * FOUT + 16 * t + l15) * NN
                            + i0 + halfL * 8;
        *(v8h*)dst = pk;   // 16B aligned (i0 mult of 16, halfL*8 mult of 8)
    }

    // s_i / d_i : reduce over features. Each lane has partial over n = 16t+l15.
    float ps[8] = {0.f,0.f,0.f,0.f,0.f,0.f,0.f,0.f};
    float pd[8] = {0.f,0.f,0.f,0.f,0.f,0.f,0.f,0.f};
    #pragma unroll
    for (int t = 0; t < 8; ++t) {
        float as = a_src[16 * t + l15];
        float ad = a_dst[16 * t + l15];
        #pragma unroll
        for (int r = 0; r < 8; ++r) {
            ps[r] += C[t][r] * as;
            pd[r] += C[t][r] * ad;
        }
    }
    // xor-reduce across the 16 lanes of each half-wave
    #pragma unroll
    for (int m = 1; m < 16; m <<= 1) {
        #pragma unroll
        for (int r = 0; r < 8; ++r) {
            ps[r] += __shfl_xor(ps[r], m, 32);
            pd[r] += __shfl_xor(pd[r], m, 32);
        }
    }
    if (l15 == 0) {
        #pragma unroll
        for (int r = 0; r < 8; ++r) {
            s_out[row0 + halfL * 8 + r]  = ps[r];
            d_out_[row0 + halfL * 8 + r] = pd[r];
        }
    }
}

// ---------------------------------------------------------------------------
// Kernel 1b: per-batch max of d (leaky_relu monotone -> row max from this)
// ---------------------------------------------------------------------------
__global__ void __launch_bounds__(256)
gat_k1b_dmax(const float* __restrict__ d, float* __restrict__ dmax)
{
    __shared__ float red[256];
    const int b = blockIdx.x;
    float m = -3.0e38f;
    for (int j = threadIdx.x; j < NN; j += 256) m = fmaxf(m, d[b * NN + j]);
    red[threadIdx.x] = m;
    __syncthreads();
    for (int sft = 128; sft > 0; sft >>= 1) {
        if (threadIdx.x < sft)
            red[threadIdx.x] = fmaxf(red[threadIdx.x], red[threadIdx.x + sft]);
        __syncthreads();
    }
    if (threadIdx.x == 0) dmax[b] = red[0];
}

// ---------------------------------------------------------------------------
// Kernel 2: softmax denominators. One thread per row; d[] staged in LDS.
// ---------------------------------------------------------------------------
__global__ void __launch_bounds__(256)
gat_k2_denom(const float* __restrict__ s, const float* __restrict__ d,
             const float* __restrict__ dmax,
             float* __restrict__ mrow, float* __restrict__ invden)
{
    __shared__ float dl[NN];  // 16 KB
    const int g0 = blockIdx.x * 256;     // 4096/256=16 blocks/batch: no straddle
    const int b  = g0 / NN;
    for (int j = threadIdx.x; j < NN; j += 256) dl[j] = d[b * NN + j];
    __syncthreads();

    const int g  = g0 + threadIdx.x;
    const float si = s[g];
    const float mi = lrelu(si + dmax[b]);
    float acc = 0.f;
    #pragma unroll 4
    for (int j = 0; j < NN; ++j)
        acc += __expf(lrelu(si + dl[j]) - mi);
    mrow[g]   = mi;
    invden[g] = 1.0f / acc;
}

// ---------------------------------------------------------------------------
// Kernel 3: fused attn materialization + (attn @ h_prime) via f16 WMMA.
// One wave = 16 output rows, C = 16x128 fp32 in 8 v8f accumulators.
// Per j-step (32 cols): each lane computes its 16 softmax probs directly in
// the A-fragment layout (M=l15, K chunks kb..kb+7 and kb+16..kb+23,
// kb=halfL*8), writes them fp32 to attn with NON-TEMPORAL stores (pure
// 537 MB streaming write; keep L2 resident for hpT), packs f16, then 8 WMMAs
// against contiguous B tiles from transposed h_prime.
// ---------------------------------------------------------------------------
__global__ void __launch_bounds__(128)
gat_k3_main(const _Float16* __restrict__ hpT, const float* __restrict__ s,
            const float* __restrict__ d, const float* __restrict__ mrow,
            const float* __restrict__ invden, const float* __restrict__ bias,
            float* __restrict__ out, float* __restrict__ attn)
{
    const int lane  = threadIdx.x & 31;
    const int wave  = threadIdx.x >> 5;
    const int l15   = lane & 15;
    const int halfL = lane >> 4;
    const int kb    = halfL * 8;

    const int tile = blockIdx.x * 4 + wave;
    const int row0 = tile * 16;
    const int b    = row0 / NN;
    const int i0   = row0 % NN;
    const int i    = i0 + l15;                 // A-fragment row for this lane
    const size_t gi = (size_t)b * NN + i;

    const float si  = s[gi];
    const float mi  = mrow[gi];
    const float inv = invden[gi];

    float* attn_row = attn + gi * NN;
    const float*    drow0 = d + (size_t)b * NN;
    const _Float16* hpTb  = hpT + (size_t)b * FOUT * NN;

    v8f C[8] = {};

    for (int j0 = 0; j0 < NN; j0 += 32) {
        // ---- attention probabilities for this lane's A fragment ----
        v16h A;
        float p0[8], p1[8];
        const float* dj = drow0 + j0 + kb;
        #pragma unroll
        for (int c = 0; c < 8; ++c) {
            float x0 = si + dj[c];
            float x1 = si + dj[c + 16];
            p0[c] = __expf(lrelu(x0) - mi) * inv;
            p1[c] = __expf(lrelu(x1) - mi) * inv;
            A[c]     = (_Float16)p0[c];
            A[c + 8] = (_Float16)p1[c];
        }
        // ---- write attn (fp32, exact softmax values), non-temporal ----
        __builtin_nontemporal_store((v4f){p0[0], p0[1], p0[2], p0[3]},
                                    (v4f*)(attn_row + j0 + kb));
        __builtin_nontemporal_store((v4f){p0[4], p0[5], p0[6], p0[7]},
                                    (v4f*)(attn_row + j0 + kb + 4));
        __builtin_nontemporal_store((v4f){p1[0], p1[1], p1[2], p1[3]},
                                    (v4f*)(attn_row + j0 + kb + 16));
        __builtin_nontemporal_store((v4f){p1[4], p1[5], p1[6], p1[7]},
                                    (v4f*)(attn_row + j0 + kb + 20));

        // ---- 8 WMMAs: B fragment = K chunks kb..kb+7 and kb+16..kb+23 (same as A) ----
        const int jb = j0 + kb;
        #pragma unroll
        for (int t = 0; t < 8; ++t) {
            const _Float16* bp = hpTb + (size_t)(16 * t + l15) * NN + jb;
            v8h blo = *(const v8h*)bp;         // 16B aligned (jb mult of 8)
            v8h bhi = *(const v8h*)(bp + 16);
            v16h B = __builtin_shufflevector(blo, bhi, 0,1,2,3,4,5,6,7,8,9,10,11,12,13,14,15);
            C[t] = __builtin_amdgcn_wmma_f32_16x16x32_f16(
                false, A, false, B, (short)0, C[t], false, false);
        }
    }

    // ---- epilogue: C[t][r] = out[i0 + r + halfL*8][16t + l15], add bias ----
    #pragma unroll
    for (int t = 0; t < 8; ++t) {
        const int f = 16 * t + l15;
        const float bv = bias[f];
        #pragma unroll
        for (int r = 0; r < 8; ++r) {
            const int row = i0 + halfL * 8 + r;
            __builtin_nontemporal_store(C[t][r] + bv,
                out + ((size_t)b * NN + row) * FOUT + f);
        }
    }
}

// ---------------------------------------------------------------------------
extern "C" void kernel_launch(void* const* d_in, const int* in_sizes, int n_in,
                              void* d_out, int out_size, void* d_ws, size_t ws_size,
                              hipStream_t stream) {
    const float* h     = (const float*)d_in[0];
    const float* w     = (const float*)d_in[1];
    const float* a_src = (const float*)d_in[2];
    const float* a_dst = (const float*)d_in[3];
    const float* bias  = (const float*)d_in[4];

    float* out  = (float*)d_out;                      // [8,4096,128]
    float* attn = out + (size_t)BS * NN * FOUT;       // [8,4096,4096]

    // workspace layout
    char* p = (char*)d_ws;
    _Float16* hpT = (_Float16*)p;  p += (size_t)BS * FOUT * NN * sizeof(_Float16); // 8 MB
    float* s      = (float*)p;     p += (size_t)BS * NN * sizeof(float);
    float* dvec   = (float*)p;     p += (size_t)BS * NN * sizeof(float);
    float* mrow   = (float*)p;     p += (size_t)BS * NN * sizeof(float);
    float* invden = (float*)p;     p += (size_t)BS * NN * sizeof(float);
    float* dmax   = (float*)p;     p += 256;

    const int tiles  = BS * NN / 16;   // 2048 sixteen-row tiles
    const int blocks = tiles / 4;      // 4 waves / block

    gat_k1_hprime<<<blocks, 128, 0, stream>>>(h, w, a_src, a_dst, hpT, s, dvec);
    gat_k1b_dmax <<<BS, 256, 0, stream>>>(dvec, dmax);
    gat_k2_denom <<<BS * NN / 256, 256, 0, stream>>>(s, dvec, dmax, mrow, invden);
    gat_k3_main  <<<blocks, 128, 0, stream>>>(hpT, s, dvec, mrow, invden, bias,
                                              out, attn);
}
